// Transformer_74491912781838
// MI455X (gfx1250) — matrix operand order repaired and checked by reading, not compiled, not run on hardware
//
#include <hip/hip_runtime.h>

#define DM 256
#define NHH 4
#define KK 1024
#define FFD 1024
#define SS 2048
#define BB 4
#define ROWS (SS*BB)   // 8192

typedef __attribute__((ext_vector_type(16))) __bf16 v16bf;
typedef __attribute__((ext_vector_type(8)))  __bf16 v8bf;
typedef __attribute__((ext_vector_type(8)))  float  v8f;
typedef __attribute__((ext_vector_type(4)))  unsigned int v4u;
typedef __attribute__((ext_vector_type(8)))  int v8i;
typedef __attribute__((ext_vector_type(4)))  int v4i;

#if __has_builtin(__builtin_amdgcn_tensor_load_to_lds)
#define TDM_PATH 1
#else
#define TDM_PATH 0
#endif

// ---------------- WMMA fragment loaders (wave32, 16x16x32 bf16) ----------------
// A (16xK tile, row-major, ld elements per row):
//   lanes 0-15 : row = lane,   K chunks {k0+0..7} and {k0+16..23}
//   lanes 16-31: row = lane-16,K chunks {k0+8..15} and {k0+24..31}
__device__ __forceinline__ v16bf ld_fragA(const __bf16* base, int ld, int k0, int lane) {
  int r  = lane & 15;
  int kb = k0 + ((lane & 16) ? 8 : 0);
  const __bf16* p = base + (long)r * ld + kb;
  v8bf lo = *(const v8bf*)p;
  v8bf hi = *(const v8bf*)(p + 16);
  v16bf f;
#pragma unroll
  for (int i = 0; i < 8; ++i) { f[i] = lo[i]; f[8 + i] = hi[i]; }
  return f;
}
// B (KxN=32x16). Source stored as N rows of K ("B^T" rows, contiguous over K):
//   lanes 0-15 : col = lane,   K chunks {k0+0..7} and {k0+16..23}
//   lanes 16-31: col = lane-16,K chunks {k0+8..15} and {k0+24..31}
__device__ __forceinline__ v16bf ld_fragB(const __bf16* base, int ld, int k0, int lane) {
  int r  = lane & 15;
  int kb = k0 + ((lane & 16) ? 8 : 0);
  const __bf16* p = base + (long)r * ld + kb;
  v8bf lo = *(const v8bf*)p;
  v8bf hi = *(const v8bf*)(p + 16);
  v16bf f;
#pragma unroll
  for (int i = 0; i < 8; ++i) { f[i] = lo[i]; f[8 + i] = hi[i]; }
  return f;
}

__device__ __forceinline__ float red_add16(float v) {
#pragma unroll
  for (int m = 1; m < 16; m <<= 1) v += __shfl_xor(v, m, 32);
  return v;
}
__device__ __forceinline__ float red_max16(float v) {
#pragma unroll
  for (int m = 1; m < 16; m <<= 1) v = fmaxf(v, __shfl_xor(v, m, 32));
  return v;
}

// ---------------- TDM: stage one 16x1024-bf16 tile (32KB, contiguous rows) into LDS --------
// 2D descriptor, data_size = 8B units: tile 256x16 units, row stride 256 units.
__device__ __forceinline__ void tdm_issue_tile(const __bf16* gsrc, unsigned int lds_byte_off) {
#if TDM_PATH
  unsigned long long ga = (unsigned long long)(const void*)gsrc;
  v4u g0;
  g0[0] = 1u;                                   // count=1, user descriptor
  g0[1] = lds_byte_off;                         // lds_addr
  g0[2] = (unsigned int)(ga & 0xffffffffu);     // global_addr[31:0]
  g0[3] = (unsigned int)((ga >> 32) & 0x1ffffffu) | (2u << 30);  // addr[56:32] | type=2
  v8i g1;
  g1[0] = (3 << 16);                            // workgroup_mask=0, data_size=3 (8B)
  g1[1] = (int)(256u << 16);                    // tensor_dim0[15:0] = 256
  g1[2] = (int)(0xffffu << 16);                 // tensor_dim0 hi=0 | tensor_dim1 lo16
  g1[3] = (int)(256u << 16);                    // tensor_dim1 hi=0 | tile_dim0 = 256
  g1[4] = 16;                                   // tile_dim1 = 16, tile_dim2 = 0
  g1[5] = 256;                                  // tensor_dim0_stride = 256 units
  g1[6] = 0;
  g1[7] = 0;
  v4i gz4 = {};
  v8i gz8 = {};
  __builtin_amdgcn_tensor_load_to_lds(g0, g1, gz4, gz4, gz8, 0);
#else
  (void)gsrc; (void)lds_byte_off;
#endif
}

__device__ __forceinline__ void coop_copy_tile(__bf16* dst, const __bf16* src, int tid) {
  // fallback: 128 threads copy 16B-vectors of the 32KB tile
  for (int i = tid; i < 16 * KK / 8; i += 128)
    ((v8bf*)dst)[i] = ((const v8bf*)src)[i];
}

// ---------------- fp32 -> bf16 convert ----------------
__global__ __launch_bounds__(256) void cvt_bf16_kernel(const float* __restrict__ in,
                                                       __bf16* __restrict__ out, long n) {
  long i = (long)blockIdx.x * blockDim.x + threadIdx.x;
  long stride = (long)gridDim.x * blockDim.x;
  for (; i < n; i += stride) out[i] = (__bf16)in[i];
}

// ---------------- generic row GEMM: OUT[h][orow][k] = act(X[row] . W[h][k] + bias[k]) -------
__global__ __launch_bounds__(128)
void gemm_rows_kernel(const __bf16* __restrict__ X, const __bf16* __restrict__ W,
                      const float* __restrict__ bias, __bf16* __restrict__ OUT,
                      int kout, long whstride, long ohstride,
                      int swizzle, int do_relu, int do_norm) {
  __shared__ __bf16 tile[16 * 1024];
  __shared__ float ssum[16];
  const int tid  = threadIdx.x;
  const int lane = tid & 31;
  const int wave = tid >> 5;
  const int rt   = blockIdx.x;
  const int h    = blockIdx.y;
  const __bf16* Xt = X + (long)rt * 16 * DM;
  const __bf16* Wh = W + (long)h * whstride;
  if (tid < 16) ssum[tid] = 0.f;
  __syncthreads();

  const int nkt = kout >> 4;
  const int col = lane & 15;
  const int rbase = (lane & 16) ? 8 : 0;
  for (int kt = wave; kt < nkt; kt += 4) {
    const __bf16* Wt2 = Wh + (long)kt * 16 * DM;
    // batch all 8 B fragments first: 16 outstanding loads, one wait per 8 WMMAs
    v16bf bf[8];
#pragma unroll
    for (int j = 0; j < 8; ++j) bf[j] = ld_fragB(Wt2, DM, j * 32, lane);
    v8f acc = {};
#pragma unroll
    for (int j = 0; j < 8; ++j) {
      v16bf a = ld_fragA(Xt, DM, j * 32, lane);
      acc = __builtin_amdgcn_wmma_f32_16x16x32_bf16(false, a, false, bf[j], (short)0, acc, false, false);
    }
    float bv = bias ? bias[kt * 16 + col] : 0.f;
#pragma unroll
    for (int i = 0; i < 8; ++i) {
      float v = acc[i] + bv;
      if (do_relu) v = fmaxf(v, 0.f);
      tile[(rbase + i) * 1024 + kt * 16 + col] = (__bf16)v;
      if (do_norm) {
        float rs = red_add16(v * v);
        if ((lane & 15) == 0) atomicAdd(&ssum[rbase + i], rs);
      }
    }
  }
  __syncthreads();
  if (do_norm && tid < 16) ssum[tid] = 1.f / fmaxf(sqrtf(ssum[tid]), 1e-12f);
  __syncthreads();

  const int kshift = (kout == 1024) ? 10 : 8;
  for (int idx = tid; idx < 16 * kout; idx += 128) {
    int r = idx >> kshift;
    int k = idx & (kout - 1);
    float v = (float)tile[r * 1024 + k];
    if (do_norm) v *= ssum[r];
    int row = rt * 16 + r;
    long orow = swizzle ? ((long)(row & 3) * SS + (row >> 2)) : (long)row;
    OUT[(long)h * ohstride + orow * kout + k] = (__bf16)v;
  }
}

// ---------------- attention pass 1: online softmax stats per query row ----------------
// wq/wk layout: [(h*B+b)][s][1024] bf16. Block = 4 waves, one q-tile each; the wk
// s-tile is shared by all waves and streamed via TDM into a double-buffered LDS region.
__global__ __launch_bounds__(128)
void attn_pass1_kernel(const __bf16* __restrict__ wq, const __bf16* __restrict__ wk,
                       float* __restrict__ mOut, float* __restrict__ lOut) {
  extern __shared__ __bf16 smem[];                  // 2 * 16 * 1024 bf16 = 64KB
  const int tid  = threadIdx.x;
  const int lane = tid & 31;
  const int wave = tid >> 5;
  const int hb   = blockIdx.x >> 5;                 // 16 (h,b) pairs
  const int qt   = (blockIdx.x & 31) * 4 + wave;    // 128 q tiles
  const __bf16* qb = wq + ((long)hb * SS + qt * 16) * KK;
  const __bf16* kb = wk + (long)hb * SS * KK;
  const unsigned int lds0 = (unsigned int)(size_t)(void*)&smem[0];

  float m[8], l[8];
#pragma unroll
  for (int i = 0; i < 8; ++i) { m[i] = -1e30f; l[i] = 0.f; }

#if TDM_PATH
  if (wave == 0) tdm_issue_tile(kb, lds0);
#else
  coop_copy_tile(&smem[0], kb, tid);
#endif

  for (int st = 0; st < 128; ++st) {
    const int cur = st & 1;
#if TDM_PATH
    if (wave == 0) __builtin_amdgcn_s_wait_tensorcnt(0);
#endif
    __syncthreads();                                // buf[cur] ready for everyone
    if (st + 1 < 128) {
#if TDM_PATH
      if (wave == 0)
        tdm_issue_tile(kb + (long)(st + 1) * 16 * KK, lds0 + (unsigned)(cur ^ 1) * 16 * KK * 2);
#else
      coop_copy_tile(&smem[(cur ^ 1) * 16 * KK], kb + (long)(st + 1) * 16 * KK, tid);
#endif
    }
    const __bf16* bt = &smem[cur * 16 * KK];
    v8f acc = {};
#pragma unroll
    for (int kc = 0; kc < 4; ++kc) {
      // batch the 8 global A fragments of this 256-wide chunk
      v16bf af[8];
#pragma unroll
      for (int j = 0; j < 8; ++j) af[j] = ld_fragA(qb, KK, kc * 256 + j * 32, lane);
#pragma unroll
      for (int j = 0; j < 8; ++j) {
        v16bf b = ld_fragB(bt, KK, kc * 256 + j * 32, lane);   // LDS, short latency
        acc = __builtin_amdgcn_wmma_f32_16x16x32_bf16(false, af[j], false, b, (short)0, acc, false, false);
      }
    }
#pragma unroll
    for (int i = 0; i < 8; ++i) {
      float tmax = red_max16(acc[i]);
      float mn = fmaxf(m[i], tmax);
      float ps = red_add16(__expf(acc[i] - mn));
      l[i] = l[i] * __expf(m[i] - mn) + ps;
      m[i] = mn;
    }
    __syncthreads();                                // everyone done with buf[cur] before reuse
  }
  if ((lane & 15) == 0) {
    int rb = (lane & 16) ? 8 : 0;
#pragma unroll
    for (int i = 0; i < 8; ++i) {
      long idx = (long)hb * SS + qt * 16 + rb + i;
      mOut[idx] = m[i];
      lOut[idx] = l[i];
    }
  }
}

// ---------------- attention pass 2: per-key-column denominator + q=0 prob row ----------------
// Block = 4 waves, one s-tile each; the wq q-tile is shared and streamed via TDM.
__global__ __launch_bounds__(128)
void attn_pass2_kernel(const __bf16* __restrict__ wq, const __bf16* __restrict__ wk,
                       const float* __restrict__ mIn, const float* __restrict__ lIn,
                       float* __restrict__ prob0, float* __restrict__ denom) {
  extern __shared__ __bf16 smem[];                  // 64KB double buffer
  const int tid  = threadIdx.x;
  const int lane = tid & 31;
  const int wave = tid >> 5;
  const int hb   = blockIdx.x >> 5;
  const int st   = (blockIdx.x & 31) * 4 + wave;
  const __bf16* ktp = wk + ((long)hb * SS + st * 16) * KK;
  const __bf16* qb0 = wq + (long)hb * SS * KK;
  const unsigned int lds0 = (unsigned int)(size_t)(void*)&smem[0];
  const int rb = (lane & 16) ? 8 : 0;
  float dcol = 0.f, p0 = 0.f;

#if TDM_PATH
  if (wave == 0) tdm_issue_tile(qb0, lds0);
#else
  coop_copy_tile(&smem[0], qb0, tid);
#endif

  for (int qt = 0; qt < 128; ++qt) {
    const int cur = qt & 1;
#if TDM_PATH
    if (wave == 0) __builtin_amdgcn_s_wait_tensorcnt(0);
#endif
    __syncthreads();
    if (qt + 1 < 128) {
#if TDM_PATH
      if (wave == 0)
        tdm_issue_tile(qb0 + (long)(qt + 1) * 16 * KK, lds0 + (unsigned)(cur ^ 1) * 16 * KK * 2);
#else
      coop_copy_tile(&smem[(cur ^ 1) * 16 * KK], qb0 + (long)(qt + 1) * 16 * KK, tid);
#endif
    }
    const __bf16* at = &smem[cur * 16 * KK];
    v8f acc = {};
#pragma unroll
    for (int kc = 0; kc < 4; ++kc) {
      // batch the 8 global B fragments of this 256-wide chunk
      v16bf bfr[8];
#pragma unroll
      for (int j = 0; j < 8; ++j) bfr[j] = ld_fragB(ktp, KK, kc * 256 + j * 32, lane);
#pragma unroll
      for (int j = 0; j < 8; ++j) {
        v16bf a = ld_fragA(at, KK, kc * 256 + j * 32, lane);   // LDS, short latency
        acc = __builtin_amdgcn_wmma_f32_16x16x32_bf16(false, a, false, bfr[j], (short)0, acc, false, false);
      }
    }
#pragma unroll
    for (int i = 0; i < 8; ++i) {
      int row = qt * 16 + rb + i;
      long sidx = (long)hb * SS + row;
      float p = __expf(acc[i] - mIn[sidx]) / lIn[sidx];
      dcol += p;
      if (row == 0) p0 = p;                         // only qt==0, lanes 0-15, i==0
    }
    __syncthreads();
  }
  dcol += __shfl_xor(dcol, 16, 32);
  if (lane < 16) {
    long idx = (long)hb * SS + st * 16 + lane;
    denom[idx] = dcol;
    prob0[idx] = p0;
  }
}

// ---------------- o[q=0] = sum_s aff0[s] * wv[h,b,s,:] ----------------
__global__ __launch_bounds__(256)
void o0_kernel(const float* __restrict__ prob0, const float* __restrict__ denom,
               const __bf16* __restrict__ wv, float* __restrict__ o0) {
  const int hb = blockIdx.x;
  const int e  = threadIdx.x;
  float a = 0.f;
  for (int s = 0; s < SS; ++s) {
    long si = (long)hb * SS + s;
    float w = prob0[si] / (1e-9f + denom[si]);
    a += w * (float)wv[si * DM + e];
  }
  o0[(long)hb * DM + e] = a;
}

// ---------------- per-head trans + TransNonlinear + LN (q=0 only) ----------------
__global__ __launch_bounds__(256)
void head_kernel(const float* __restrict__ memory, const float* __restrict__ o0,
                 const float* __restrict__ Wt, const float* __restrict__ e1w,
                 const float* __restrict__ e1b, const float* __restrict__ e2w,
                 const float* __restrict__ e2b, const float* __restrict__ eg,
                 const float* __restrict__ eb, float* __restrict__ xh) {
  __shared__ float sdiff[DM];
  __shared__ float sru[DM];
  __shared__ float shmid[KK];
  __shared__ float sred[2];
  const int hb = blockIdx.x, h = hb >> 2, b = hb & 3, t = threadIdx.x;
  sdiff[t] = memory[b * DM + t] - o0[(long)hb * DM + t];  // memory[s=0,b,:]
  __syncthreads();
  {
    const float* w = Wt + ((long)h * DM + t) * DM;
    float a = 0.f;
#pragma unroll 8
    for (int d = 0; d < DM; ++d) a += sdiff[d] * w[d];
    sru[t] = fmaxf(a, 0.f);
  }
  __syncthreads();
  for (int j = 0; j < 4; ++j) {
    int k = t + j * 256;
    const float* w = e1w + ((long)h * KK + k) * DM;
    float a = e1b[h * KK + k];
#pragma unroll 8
    for (int d = 0; d < DM; ++d) a += sru[d] * w[d];
    shmid[k] = fmaxf(a, 0.f);
  }
  __syncthreads();
  float v;
  {
    const float* w = e2w + ((long)h * DM + t) * KK;
    float a = e2b[h * DM + t];
#pragma unroll 8
    for (int k = 0; k < KK; ++k) a += shmid[k] * w[k];
    v = sru[t] + a;
  }
  if (t < 2) sred[t] = 0.f;
  __syncthreads();
  atomicAdd(&sred[0], v);
  atomicAdd(&sred[1], v * v);
  __syncthreads();
  float mean = sred[0] * (1.f / DM);
  float var  = sred[1] * (1.f / DM) - mean * mean;
  float inv  = rsqrtf(var + 1e-5f);
  xh[(long)hb * DM + t] = (v - mean) * inv * eg[h * DM + t] + eb[h * DM + t];
}

// ---------------- srcc2[b,d] = cat(xh) . Wout[d,:] ----------------
__global__ __launch_bounds__(256)
void rma0_kernel(const float* __restrict__ xh, const float* __restrict__ Wout,
                 float* __restrict__ srcc2) {
  const int d = threadIdx.x;
  for (int b = 0; b < BB; ++b) {
    float a = 0.f;
#pragma unroll 8
    for (int j = 0; j < NHH * DM; ++j) {
      int h = j >> 8, e = j & 255;
      a += xh[((long)(h * BB + b)) * DM + e] * Wout[(long)d * (NHH * DM) + j];
    }
    srcc2[b * DM + d] = a;
  }
}

// ---------------- x = LN(memory + srcc2) ; stash fp32 + bf16 ----------------
__global__ __launch_bounds__(256)
void ln_x_kernel(const float* __restrict__ memory, const float* __restrict__ srcc2,
                 const float* __restrict__ n1g, const float* __restrict__ n1b,
                 float* __restrict__ xrow, __bf16* __restrict__ Xf) {
  __shared__ float sred[2];
  const int row = blockIdx.x, b = row & 3, t = threadIdx.x;
  float v = memory[(long)row * DM + t] + srcc2[b * DM + t];
  if (t < 2) sred[t] = 0.f;
  __syncthreads();
  atomicAdd(&sred[0], v);
  atomicAdd(&sred[1], v * v);
  __syncthreads();
  float mean = sred[0] * (1.f / DM);
  float var  = sred[1] * (1.f / DM) - mean * mean;
  float inv  = rsqrtf(var + 1e-5f);
  float x = (v - mean) * inv * n1g[t] + n1b[t];
  xrow[(long)row * DM + t] = x;
  Xf[(long)row * DM + t]   = (__bf16)x;
}

// ---------------- FFN second GEMM + residual + LN -> d_out ----------------
__global__ __launch_bounds__(128)
void ffn2_ln_kernel(const __bf16* __restrict__ H, const __bf16* __restrict__ W2,
                    const float* __restrict__ l2b, const float* __restrict__ xrow,
                    const float* __restrict__ n3g, const float* __restrict__ n3b,
                    float* __restrict__ out) {
  __shared__ float tile[16 * DM];
  __shared__ float red[32];
  const int tid = threadIdx.x, lane = tid & 31, wave = tid >> 5;
  const int rt = blockIdx.x;
  const __bf16* Ht = H + (long)rt * 16 * FFD;
  const int col = lane & 15;
  const int rbase = (lane & 16) ? 8 : 0;
  for (int kt = wave; kt < 16; kt += 4) {
    const __bf16* Wt2 = W2 + (long)kt * 16 * FFD;
    v8f acc = {};
#pragma unroll
    for (int kc = 0; kc < 4; ++kc) {
      v16bf bfr[8];
#pragma unroll
      for (int j = 0; j < 8; ++j) bfr[j] = ld_fragB(Wt2, FFD, kc * 256 + j * 32, lane);
#pragma unroll
      for (int j = 0; j < 8; ++j) {
        v16bf a = ld_fragA(Ht, FFD, kc * 256 + j * 32, lane);
        acc = __builtin_amdgcn_wmma_f32_16x16x32_bf16(false, a, false, bfr[j], (short)0, acc, false, false);
      }
    }
    float bv = l2b[kt * 16 + col];
#pragma unroll
    for (int i = 0; i < 8; ++i) tile[(rbase + i) * DM + kt * 16 + col] = acc[i] + bv;
  }
  if (tid < 32) red[tid] = 0.f;
  __syncthreads();
  const int r = tid >> 3;
  const int seg = (tid & 7) * 32;
  float s1 = 0.f, s2 = 0.f;
#pragma unroll 4
  for (int i = 0; i < 32; ++i) {
    int k = seg + i;
    float v = tile[r * DM + k] + xrow[((long)rt * 16 + r) * DM + k];
    tile[r * DM + k] = v;
    s1 += v;
    s2 += v * v;
  }
  atomicAdd(&red[r], s1);
  atomicAdd(&red[16 + r], s2);
  __syncthreads();
  float mean = red[r] * (1.f / DM);
  float var  = red[16 + r] * (1.f / DM) - mean * mean;
  float inv  = rsqrtf(var + 1e-5f);
#pragma unroll 4
  for (int i = 0; i < 32; ++i) {
    int k = seg + i;
    float v = (tile[r * DM + k] - mean) * inv * n3g[k] + n3b[k];
    out[((long)rt * 16 + r) * DM + k] = v;
  }
}

// ---------------- host side ----------------
static inline char* wsalloc(void* ws, size_t& off, size_t bytes) {
  char* p = (char*)ws + off;
  off += (bytes + 255) & ~size_t(255);
  return p;
}

extern "C" void kernel_launch(void* const* d_in, const int* in_sizes, int n_in,
                              void* d_out, int out_size, void* d_ws, size_t ws_size,
                              hipStream_t stream) {
  (void)in_sizes; (void)n_in; (void)out_size; (void)ws_size;
  const float* srcc   = (const float*)d_in[0];
  const float* memory = (const float*)d_in[1];
  const float* WK     = (const float*)d_in[2];
  const float* WQ     = (const float*)d_in[3];
  const float* WV     = (const float*)d_in[4];
  const float* Wt     = (const float*)d_in[5];
  const float* e1w    = (const float*)d_in[6];
  const float* e1b    = (const float*)d_in[7];
  const float* e2w    = (const float*)d_in[8];
  const float* e2b    = (const float*)d_in[9];
  const float* eg     = (const float*)d_in[10];
  const float* eb     = (const float*)d_in[11];
  const float* Wout   = (const float*)d_in[12];
  const float* l1w    = (const float*)d_in[13];
  const float* l1b    = (const float*)d_in[14];
  const float* l2w    = (const float*)d_in[15];
  const float* l2b    = (const float*)d_in[16];
  const float* n1g    = (const float*)d_in[17];
  const float* n1b    = (const float*)d_in[18];
  const float* n3g    = (const float*)d_in[19];
  const float* n3b    = (const float*)d_in[20];
  float* out = (float*)d_out;

  size_t off = 0;
  __bf16* srcb  = (__bf16*)wsalloc(d_ws, off, (size_t)ROWS * DM * 2);
  __bf16* memb  = (__bf16*)wsalloc(d_ws, off, (size_t)ROWS * DM * 2);
  __bf16* WQb   = (__bf16*)wsalloc(d_ws, off, (size_t)NHH * KK * DM * 2);
  __bf16* WKb   = (__bf16*)wsalloc(d_ws, off, (size_t)NHH * KK * DM * 2);
  __bf16* WVb   = (__bf16*)wsalloc(d_ws, off, (size_t)NHH * DM * DM * 2);
  __bf16* l1wb  = (__bf16*)wsalloc(d_ws, off, (size_t)FFD * DM * 2);
  __bf16* l2wb  = (__bf16*)wsalloc(d_ws, off, (size_t)DM * FFD * 2);
  __bf16* wqb   = (__bf16*)wsalloc(d_ws, off, (size_t)NHH * BB * SS * KK * 2);
  __bf16* wkb   = (__bf16*)wsalloc(d_ws, off, (size_t)NHH * BB * SS * KK * 2);
  __bf16* wvb   = (__bf16*)wsalloc(d_ws, off, (size_t)NHH * BB * SS * DM * 2);
  float*  mS    = (float*)wsalloc(d_ws, off, (size_t)NHH * BB * SS * 4);
  float*  lS    = (float*)wsalloc(d_ws, off, (size_t)NHH * BB * SS * 4);
  float*  prob0 = (float*)wsalloc(d_ws, off, (size_t)NHH * BB * SS * 4);
  float*  denom = (float*)wsalloc(d_ws, off, (size_t)NHH * BB * SS * 4);
  float*  o0    = (float*)wsalloc(d_ws, off, (size_t)NHH * BB * DM * 4);
  float*  xh    = (float*)wsalloc(d_ws, off, (size_t)NHH * BB * DM * 4);
  float*  srcc2 = (float*)wsalloc(d_ws, off, (size_t)BB * DM * 4);
  float*  xrow  = (float*)wsalloc(d_ws, off, (size_t)ROWS * DM * 4);
  __bf16* Xf    = (__bf16*)wsalloc(d_ws, off, (size_t)ROWS * DM * 2);
  __bf16* Hb    = (__bf16*)wsalloc(d_ws, off, (size_t)ROWS * FFD * 2);

  auto cvt = [&](const float* in, __bf16* o, long n) {
    int blocks = (int)((n + 255) / 256);
    if (blocks > 4096) blocks = 4096;
    hipLaunchKernelGGL(cvt_bf16_kernel, dim3(blocks), dim3(256), 0, stream, in, o, n);
  };
  cvt(srcc, srcb, (long)ROWS * DM);
  cvt(memory, memb, (long)ROWS * DM);
  cvt(WQ, WQb, (long)NHH * KK * DM);
  cvt(WK, WKb, (long)NHH * KK * DM);
  cvt(WV, WVb, (long)NHH * DM * DM);
  cvt(l1w, l1wb, (long)FFD * DM);
  cvt(l2w, l2wb, (long)DM * FFD);

  // wq / wk projections with L2-norm, scattered to [h][b][s][k]
  hipLaunchKernelGGL(gemm_rows_kernel, dim3(ROWS / 16, NHH), dim3(128), 0, stream,
                     memb, WQb, (const float*)nullptr, wqb,
                     KK, (long)KK * DM, (long)BB * SS * KK, 1, 0, 1);
  hipLaunchKernelGGL(gemm_rows_kernel, dim3(ROWS / 16, NHH), dim3(128), 0, stream,
                     srcb, WKb, (const float*)nullptr, wkb,
                     KK, (long)KK * DM, (long)BB * SS * KK, 1, 0, 1);
  // wv projection
  hipLaunchKernelGGL(gemm_rows_kernel, dim3(ROWS / 16, NHH), dim3(128), 0, stream,
                     srcb, WVb, (const float*)nullptr, wvb,
                     DM, (long)DM * DM, (long)BB * SS * DM, 1, 0, 0);
  // attention: stats pass then denominator/prob0 pass (64KB dynamic LDS double buffer)
  hipLaunchKernelGGL(attn_pass1_kernel, dim3(512), dim3(128), 65536, stream, wqb, wkb, mS, lS);
  hipLaunchKernelGGL(attn_pass2_kernel, dim3(512), dim3(128), 65536, stream, wqb, wkb, mS, lS, prob0, denom);
  // q=0 tail
  hipLaunchKernelGGL(o0_kernel, dim3(NHH * BB), dim3(DM), 0, stream, prob0, denom, wvb, o0);
  hipLaunchKernelGGL(head_kernel, dim3(NHH * BB), dim3(DM), 0, stream,
                     memory, o0, Wt, e1w, e1b, e2w, e2b, eg, eb, xh);
  hipLaunchKernelGGL(rma0_kernel, dim3(1), dim3(DM), 0, stream, xh, Wout, srcc2);
  // decoder layer
  hipLaunchKernelGGL(ln_x_kernel, dim3(ROWS), dim3(DM), 0, stream, memory, srcc2, n1g, n1b, xrow, Xf);
  hipLaunchKernelGGL(gemm_rows_kernel, dim3(ROWS / 16, 1), dim3(128), 0, stream,
                     Xf, l1wb, l1b, Hb, FFD, 0L, 0L, 0, 1, 0);
  hipLaunchKernelGGL(ffn2_ln_kernel, dim3(ROWS / 16), dim3(128), 0, stream,
                     Hb, l2wb, l2b, xrow, n3g, n3b, out);
}
